// Encoder3DJEPA_21698174779769
// MI455X (gfx1250) — matrix operand order repaired and checked by reading, not compiled, not run on hardware
//
#include <hip/hip_runtime.h>
#include <hip/hip_bf16.h>

typedef _Float16 half_t;
typedef __attribute__((ext_vector_type(16))) _Float16 v16h;
typedef __attribute__((ext_vector_type(8)))  _Float16 v8h;
typedef __attribute__((ext_vector_type(8)))  float    v8f;

// ---------------- problem constants ----------------
constexpr int cB    = 2;
constexpr int cN    = 40960;
constexpr int cFD   = 512;
constexpr int cRP   = 128;
constexpr int cD    = 768;
constexpr int cPS   = 128;
constexpr int cH    = 12;
constexpr int cL    = 2;
constexpr int cM    = cB * cN;        // 81920 tokens total
constexpr int cSORT = 65536;          // padded sort length per batch

// GEMM LDS strides (halves); *2 bytes must be multiple of 16 for b128 access
constexpr int GLD = 40;

// attention LDS strides (halves)
constexpr int LQ = 72, LK = 72, LV = 136, LP = 136;
constexpr int ATT_Q_OFF = 0;
constexpr int ATT_K_OFF = ATT_Q_OFF + 128 * LQ * 2;          // 18432
constexpr int ATT_V_OFF = ATT_K_OFF + 128 * LK * 2;          // 36864
constexpr int ATT_S_OFF = ATT_V_OFF + 64  * LV * 2;          // 54272
constexpr int ATT_P_OFF = ATT_S_OFF + 128 * 128 * 4;         // 119808
constexpr int ATTN_SMEM = ATT_P_OFF + 128 * LP * 2;          // 154624 bytes

// ---------------- small device helpers ----------------
__device__ inline float gelu_f(float x) {
  return 0.5f * x * (1.0f + erff(x * 0.70710678118654752f));
}

__device__ inline float blk_sum(float v, float* sbuf) {
  int t = threadIdx.x;
  sbuf[t] = v;
  __syncthreads();
  for (int s = blockDim.x >> 1; s > 0; s >>= 1) {
    if (t < s) sbuf[t] += sbuf[t + s];
    __syncthreads();
  }
  float r = sbuf[0];
  __syncthreads();
  return r;
}

__device__ inline unsigned part1by2(unsigned x) {
  x &= 1023u;
  x = (x | x << 16) & 50331903u;
  x = (x | x << 8)  & 50393103u;
  x = (x | x << 4)  & 51130563u;
  x = (x | x << 2)  & 153391689u;
  return x;
}

// async global->LDS copy of 16 bytes (ASYNCcnt-tracked, bypasses VGPRs)
__device__ inline void async_copy16(unsigned lds_off, unsigned long long gaddr) {
  asm volatile("global_load_async_to_lds_b128 %0, %1, off"
               :: "v"(lds_off), "v"(gaddr) : "memory");
}
__device__ inline void async_wait0() {
  asm volatile("s_wait_asynccnt 0" ::: "memory");
}

// ---- WMMA fragment loaders (CDNA5 16x16x32 f16 layouts, ISA 7.12.2) ----
// A fragment: 16x32 tile, row-major in LDS with stride ld (halves).
// lane m = lane&15, ksel = lane>>4; VGPR0..3 hold K=ksel*8+0..7, VGPR4..7 hold K=16+ksel*8+0..7
__device__ inline v16h frag_A(const half_t* base, int ld, int lane) {
  int m = lane & 15, ks = lane >> 4;
  const half_t* p = base + m * ld + ks * 8;
  v8h lo = *(const v8h*)(p);
  v8h hi = *(const v8h*)(p + 16);
  v16h r;
#pragma unroll
  for (int i = 0; i < 8; ++i) { r[i] = lo[i]; r[i + 8] = hi[i]; }
  return r;
}

// B fragment: 32x16 tile stored K-contiguous per column: base[n*ld + k].
// lane n = lane&15, ksel = lane>>4; halves 0..7 -> K = ksel*8 + e, halves 8..15 -> K = 16 + ksel*8 + e
__device__ inline v16h frag_B(const half_t* base, int ld, int lane) {
  int n = lane & 15, ks = lane >> 4;
  const half_t* p = base + n * ld + ks * 8;
  v8h lo = *(const v8h*)(p);
  v8h hi = *(const v8h*)(p + 16);
  v16h r;
#pragma unroll
  for (int i = 0; i < 8; ++i) { r[i] = lo[i]; r[i + 8] = hi[i]; }
  return r;
}

// ---------------- weight conversion (f32 -> f16, transposed to [N][K]) -----
__global__ __launch_bounds__(256)
void cvt_f16_t_kernel(const float* __restrict__ s, half_t* __restrict__ d,
                      int K, int N) {
  __shared__ float tile[32][33];
  int k0 = blockIdx.y * 32, n0 = blockIdx.x * 32;
  int tx = threadIdx.x & 31, ty = threadIdx.x >> 5;   // 32 x 8
  for (int r = ty; r < 32; r += 8)
    tile[r][tx] = s[(long)(k0 + r) * N + n0 + tx];
  __syncthreads();
  for (int r = ty; r < 32; r += 8)
    d[(long)(n0 + r) * K + k0 + tx] = (half_t)tile[tx][r];
}

// rp_w (99x128) -> transposed + zero padded f16 [128][128]
__global__ void cvt_rp_pad_t_kernel(const float* __restrict__ s, half_t* __restrict__ d) {
  int n = blockIdx.x, k = threadIdx.x;
  d[n * 128 + k] = (k < 99) ? (half_t)s[k * 128 + n] : (half_t)0.f;
}

// ---------------- preprocessing ----------------
// zero-token select + LN(512) -> cat16 columns [128,640)
__global__ __launch_bounds__(256)
void prep_feats_kernel(const float* __restrict__ feats, const float* __restrict__ zt,
                       const float* __restrict__ g, const float* __restrict__ b,
                       half_t* __restrict__ cat) {
  __shared__ float sbuf[256];
  long row = blockIdx.x;
  const float* f = feats + row * (long)cFD;
  int t = threadIdx.x;
  float sa = fabsf(f[t]) + fabsf(f[t + 256]);
  sa = blk_sum(sa, sbuf);
  bool zl = (sa == 0.0f);
  float v0 = zl ? zt[t] : f[t];
  float v1 = zl ? zt[t + 256] : f[t + 256];
  float mean = blk_sum(v0 + v1, sbuf) * (1.0f / 512.0f);
  float d0 = v0 - mean, d1 = v1 - mean;
  float var = blk_sum(d0 * d0 + d1 * d1, sbuf) * (1.0f / 512.0f);
  float inv = rsqrtf(var + 1e-5f);
  half_t* orow = cat + row * (long)(cRP + cFD) + cRP;
  orow[t]       = (half_t)(d0 * inv * g[t] + b[t]);
  orow[t + 256] = (half_t)(d1 * inv * g[t + 256] + b[t + 256]);
}

// harmonic embedding (99) + LN(99) -> hrm16 (stride 128, zero padded)
__global__ __launch_bounds__(128)
void prep_hrm_kernel(const float* __restrict__ rgb, const float* __restrict__ g,
                     const float* __restrict__ b, half_t* __restrict__ out) {
  __shared__ float sbuf[128];
  long row = blockIdx.x;
  int c = threadIdx.x;
  float val = 0.f;
  if (c < 99) {
    if (c < 96) {
      int j = (c < 48) ? c : (c - 48);
      float x = rgb[row * 3 + (j >> 4)] * 255.0f;
      float e = x * exp2f((float)(j & 15));
      val = (c < 48) ? sinf(e) : cosf(e);
    } else {
      val = rgb[row * 3 + (c - 96)] * 255.0f;
    }
  }
  float mean = blk_sum((c < 99) ? val : 0.f, sbuf) * (1.0f / 99.0f);
  float d0 = (c < 99) ? (val - mean) : 0.f;
  float var = blk_sum(d0 * d0, sbuf) * (1.0f / 99.0f);
  float inv = rsqrtf(var + 1e-5f);
  out[row * 128 + c] = (c < 99) ? (half_t)(d0 * inv * g[c] + b[c]) : (half_t)0.f;
}

// ---------------- layernorm kernels ----------------
__global__ __launch_bounds__(256)
void ln_f16_kernel(const float* __restrict__ x, int C, int Cpad,
                   const float* __restrict__ g, const float* __restrict__ b,
                   half_t* __restrict__ out, int ldo, int coff) {
  __shared__ float sbuf[256];
  long row = blockIdx.x;
  const float* xr = x + row * (long)C;
  float s = 0.f;
  for (int c = threadIdx.x; c < C; c += 256) s += xr[c];
  float mean = blk_sum(s, sbuf) / (float)C;
  float s2 = 0.f;
  for (int c = threadIdx.x; c < C; c += 256) { float d = xr[c] - mean; s2 += d * d; }
  float var = blk_sum(s2, sbuf) / (float)C;
  float inv = rsqrtf(var + 1e-5f);
  half_t* orow = out + row * (long)ldo + coff;
  for (int c = threadIdx.x; c < C; c += 256)
    orow[c] = (half_t)((xr[c] - mean) * inv * g[c] + b[c]);
  for (int c = C + threadIdx.x; c < Cpad; c += 256) orow[c] = (half_t)0.f;
}

__global__ __launch_bounds__(256)
void ln_f32_kernel(const float* __restrict__ x, int C,
                   const float* __restrict__ g, const float* __restrict__ b,
                   float* __restrict__ out) {
  __shared__ float sbuf[256];
  long row = blockIdx.x;
  const float* xr = x + row * (long)C;
  float s = 0.f;
  for (int c = threadIdx.x; c < C; c += 256) s += xr[c];
  float mean = blk_sum(s, sbuf) / (float)C;
  float s2 = 0.f;
  for (int c = threadIdx.x; c < C; c += 256) { float d = xr[c] - mean; s2 += d * d; }
  float var = blk_sum(s2, sbuf) / (float)C;
  float inv = rsqrtf(var + 1e-5f);
  float* orow = out + row * (long)C;
  for (int c = threadIdx.x; c < C; c += 256)
    orow[c] = (xr[c] - mean) * inv * g[c] + b[c];
}

// final LN + scatter through sort order (avoids materializing the inverse perm)
__global__ __launch_bounds__(256)
void ln_scatter_kernel(const float* __restrict__ xs, const float* __restrict__ g,
                       const float* __restrict__ b,
                       const unsigned long long* __restrict__ keys,
                       float* __restrict__ out) {
  __shared__ float sbuf[256];
  long gi = blockIdx.x;
  long bb = gi / cN, ii = gi % cN;
  unsigned ord = (unsigned)(keys[bb * (long)cSORT + ii] & 0xFFFFFFFFull);
  const float* xr = xs + gi * (long)cD;
  float s = 0.f;
  for (int c = threadIdx.x; c < cD; c += 256) s += xr[c];
  float mean = blk_sum(s, sbuf) * (1.0f / cD);
  float s2 = 0.f;
  for (int c = threadIdx.x; c < cD; c += 256) { float d = xr[c] - mean; s2 += d * d; }
  float var = blk_sum(s2, sbuf) * (1.0f / cD);
  float inv = rsqrtf(var + 1e-5f);
  float* orow = out + (bb * (long)cN + ord) * (long)cD;
  for (int c = threadIdx.x; c < cD; c += 256)
    orow[c] = (xr[c] - mean) * inv * g[c] + b[c];
}

// ---------------- morton sort ----------------
__global__ void init_min_kernel(int* gmin) {
  if (threadIdx.x < cB * 3) gmin[threadIdx.x] = 2147483647;
}

__global__ void vox_min_kernel(const float* __restrict__ pts, int* gmin) {
  long i = (long)blockIdx.x * 256 + threadIdx.x;
  if (i >= (long)cM) return;
  long bb = i / cN;
#pragma unroll
  for (int d = 0; d < 3; ++d) {
    int gv = (int)floorf(pts[i * 3 + d] / 0.05f);
    atomicMin(&gmin[bb * 3 + d], gv);
  }
}

__global__ void make_keys_kernel(const float* __restrict__ pts, const int* __restrict__ gmin,
                                 unsigned long long* __restrict__ keys) {
  int i = blockIdx.x * 256 + threadIdx.x;
  int bb = blockIdx.y;
  unsigned long long key;
  if (i < cN) {
    long pi = ((long)bb * cN + i) * 3;
    unsigned gx = (unsigned)((int)floorf(pts[pi + 0] / 0.05f) - gmin[bb * 3 + 0]);
    unsigned gy = (unsigned)((int)floorf(pts[pi + 1] / 0.05f) - gmin[bb * 3 + 1]);
    unsigned gz = (unsigned)((int)floorf(pts[pi + 2] / 0.05f) - gmin[bb * 3 + 2]);
    unsigned code = part1by2(gx) | (part1by2(gy) << 1) | (part1by2(gz) << 2);
    key = ((unsigned long long)code << 32) | (unsigned)i;   // stable: tie-break on index
  } else {
    key = ~0ULL;
  }
  keys[(long)bb * cSORT + i] = key;
}

__global__ void bitonic_pass_kernel(unsigned long long* __restrict__ keys, int k, int j) {
  int i = blockIdx.x * 256 + threadIdx.x;
  unsigned long long* kb = keys + (long)blockIdx.y * cSORT;
  int ixj = i ^ j;
  if (ixj > i) {
    unsigned long long a = kb[i], c = kb[ixj];
    bool asc = (i & k) == 0;
    bool sw = asc ? (a > c) : (a < c);
    if (sw) { kb[i] = c; kb[ixj] = a; }
  }
}

__global__ __launch_bounds__(256)
void gather_kernel(const float* __restrict__ src, float* __restrict__ dst,
                   const unsigned long long* __restrict__ keys) {
  long gi = blockIdx.x;
  long bb = gi / cN, ii = gi % cN;
  unsigned ord = (unsigned)(keys[bb * (long)cSORT + ii] & 0xFFFFFFFFull);
  const float* s = src + (bb * (long)cN + ord) * (long)cD;
  float* d = dst + gi * (long)cD;
  for (int c = threadIdx.x; c < cD; c += 256) d[c] = s[c];
}

// ---- WMMA GEMM: C[MxN] = act(A[MxK] * B[KxN] + bias) + residual -----------
// B is supplied PRE-TRANSPOSED: Bt_g is [N][K] f16, so both tiles are
// k-contiguous and staged with async global->LDS b128 copies (no VGPR round
// trip, no scatter ds stores). 128 threads (4 waves), 64x64 tile, K step 32.
__global__ __launch_bounds__(128)
void gemm64_kernel(const half_t* __restrict__ A, const half_t* __restrict__ Bt_g,
                   const float* __restrict__ bias, const float* __restrict__ res,
                   float* __restrict__ Cf, half_t* __restrict__ Ch,
                   int Mx, int Nx, int Kx, int act) {
  __shared__ half_t As[64 * GLD];
  __shared__ half_t Bt[64 * GLD];
  int tid  = threadIdx.x;
  int wv   = tid >> 5;
  int lane = tid & 31;
  int m0 = blockIdx.y * 64;
  int n0 = blockIdx.x * 64;

  v8f acc[4];
#pragma unroll
  for (int j = 0; j < 4; ++j)
#pragma unroll
    for (int i = 0; i < 8; ++i) acc[j][i] = 0.0f;

  int arow = tid >> 1, akoff = (tid & 1) * 16;   // 64 rows x 32 k
  // loop-invariant LDS destinations (byte offsets; generic ptr low 32 = LDS off)
  unsigned ldsA = (unsigned)(unsigned long long)(As + arow * GLD + akoff);
  unsigned ldsB = (unsigned)(unsigned long long)(Bt + arow * GLD + akoff);

  const half_t* gpA = A    + (long)(m0 + arow) * Kx + akoff;
  const half_t* gpB = Bt_g + (long)(n0 + arow) * Kx + akoff;

  for (int k0 = 0; k0 < Kx; k0 += 32) {
    // async-stage A tile (64x32) and B^T tile (64n x 32k), 32B per thread each
    async_copy16(ldsA,       (unsigned long long)(gpA));
    async_copy16(ldsA + 16u, (unsigned long long)(gpA + 8));
    async_copy16(ldsB,       (unsigned long long)(gpB));
    async_copy16(ldsB + 16u, (unsigned long long)(gpB + 8));
    if (k0 + 32 < Kx) {
      __builtin_prefetch(gpA + 32, 0, 1);
      __builtin_prefetch(gpB + 32, 0, 1);
    }
    gpA += 32;
    gpB += 32;
    async_wait0();
    __syncthreads();

    v16h af = frag_A(As + wv * 16 * GLD, GLD, lane);
#pragma unroll
    for (int j = 0; j < 4; ++j) {
      v16h bf = frag_B(Bt + j * 16 * GLD, GLD, lane);
      acc[j] = __builtin_amdgcn_wmma_f32_16x16x32_f16(
          false, af, false, bf, (short)0, acc[j], false, false);
    }
    __syncthreads();
  }

  int nl = lane & 15;
  int mbase = m0 + wv * 16 + (lane >> 4) * 8;
#pragma unroll
  for (int j = 0; j < 4; ++j) {
    int n = n0 + j * 16 + nl;
    float bv = bias ? bias[n] : 0.0f;
#pragma unroll
    for (int r = 0; r < 8; ++r) {
      long m = mbase + r;
      float v = acc[j][r] + bv;
      if (act == 1) v = gelu_f(v);
      long off = m * (long)Nx + n;
      if (res) v += res[off];
      if (Cf) Cf[off] = v;
      if (Ch) Ch[off] = (half_t)v;
    }
  }
}

// ---------------- windowed attention (one WG per window x head) -------------
// qkv: f16 [token][3*768], layout which*768 + h*64 + d; o: f16 [token][h*64+d]
__global__ __launch_bounds__(256)
void attn_kernel(const half_t* __restrict__ qkv, half_t* __restrict__ o) {
  extern __shared__ char smem[];
  half_t* qL = (half_t*)(smem + ATT_Q_OFF);
  half_t* kL = (half_t*)(smem + ATT_K_OFF);
  half_t* vT = (half_t*)(smem + ATT_V_OFF);
  float*  S  = (float*) (smem + ATT_S_OFF);
  half_t* P  = (half_t*)(smem + ATT_P_OFF);

  int head = blockIdx.x;
  long t0  = (long)blockIdx.y * cPS;
  int tid  = threadIdx.x;
  int wv   = tid >> 5, lane = tid & 31;

  // load q, k row-major (async b128); v transposed manually (vT[d][p])
  {
    int row = tid >> 1, hh = (tid & 1) * 32;
    const half_t* gq = qkv + (t0 + row) * (long)(3 * cD) + head * 64 + hh;
    const half_t* gk = gq + cD;
    const half_t* gv = gq + 2 * cD;
    unsigned lq = (unsigned)(unsigned long long)(qL + row * LQ + hh);
    unsigned lk = (unsigned)(unsigned long long)(kL + row * LK + hh);
    async_copy16(lq,       (unsigned long long)(gq));
    async_copy16(lq + 16u, (unsigned long long)(gq + 8));
    async_copy16(lq + 32u, (unsigned long long)(gq + 16));
    async_copy16(lq + 48u, (unsigned long long)(gq + 24));
    async_copy16(lk,       (unsigned long long)(gk));
    async_copy16(lk + 16u, (unsigned long long)(gk + 8));
    async_copy16(lk + 32u, (unsigned long long)(gk + 16));
    async_copy16(lk + 48u, (unsigned long long)(gk + 24));
    v8h v0 = *(const v8h*)(gv),      v1 = *(const v8h*)(gv + 8);
    v8h v2 = *(const v8h*)(gv + 16), v3 = *(const v8h*)(gv + 24);
#pragma unroll
    for (int e = 0; e < 8; ++e) {
      vT[(hh + e)      * LV + row] = v0[e];
      vT[(hh + 8 + e)  * LV + row] = v1[e];
      vT[(hh + 16 + e) * LV + row] = v2[e];
      vT[(hh + 24 + e) * LV + row] = v3[e];
    }
    async_wait0();
  }
  __syncthreads();

  // S = q @ k^T : each wave computes a 16x128 strip
  v8f sacc[8];
#pragma unroll
  for (int j = 0; j < 8; ++j)
#pragma unroll
    for (int i = 0; i < 8; ++i) sacc[j][i] = 0.0f;
  for (int k0 = 0; k0 < 64; k0 += 32) {
    v16h af = frag_A(qL + wv * 16 * LQ + k0, LQ, lane);
#pragma unroll
    for (int j = 0; j < 8; ++j) {
      v16h bf = frag_B(kL + j * 16 * LK + k0, LK, lane);
      sacc[j] = __builtin_amdgcn_wmma_f32_16x16x32_f16(
          false, af, false, bf, (short)0, sacc[j], false, false);
    }
  }
  int nl = lane & 15;
  int mloc = wv * 16 + (lane >> 4) * 8;
#pragma unroll
  for (int j = 0; j < 8; ++j)
#pragma unroll
    for (int r = 0; r < 8; ++r)
      S[(mloc + r) * 128 + j * 16 + nl] = sacc[j][r] * 0.125f;
  __syncthreads();

  // softmax per row -> P (f16)
  if (tid < 128) {
    float* srow = S + tid * 128;
    float mx = -1e30f;
    for (int c = 0; c < 128; ++c) mx = fmaxf(mx, srow[c]);
    float sum = 0.f;
    for (int c = 0; c < 128; ++c) { float e = __expf(srow[c] - mx); srow[c] = e; sum += e; }
    float rinv = 1.0f / sum;
    half_t* prow = P + tid * LP;
    for (int c = 0; c < 128; ++c) prow[c] = (half_t)(srow[c] * rinv);
  }
  __syncthreads();

  // O = P @ V : each wave computes a 16x64 strip
  v8f oacc[4];
#pragma unroll
  for (int j = 0; j < 4; ++j)
#pragma unroll
    for (int i = 0; i < 8; ++i) oacc[j][i] = 0.0f;
  for (int k0 = 0; k0 < 128; k0 += 32) {
    v16h af = frag_A(P + wv * 16 * LP + k0, LP, lane);
#pragma unroll
    for (int j = 0; j < 4; ++j) {
      v16h bf = frag_B(vT + j * 16 * LV + k0, LV, lane);
      oacc[j] = __builtin_amdgcn_wmma_f32_16x16x32_f16(
          false, af, false, bf, (short)0, oacc[j], false, false);
    }
  }
#pragma unroll
  for (int j = 0; j < 4; ++j)
#pragma unroll
    for (int r = 0; r < 8; ++r) {
      int m = mloc + r;
      int n = j * 16 + nl;
      o[(t0 + m) * (long)cD + head * 64 + n] = (half_t)oacc[j][r];
    }
}

// ---------------- host orchestration ----------------
extern "C" void kernel_launch(void* const* d_in, const int* in_sizes, int n_in,
                              void* d_out, int out_size, void* d_ws, size_t ws_size,
                              hipStream_t stream) {
  (void)in_sizes; (void)n_in; (void)out_size; (void)ws_size;
  const float* features = (const float*)d_in[0];
  const float* rgb      = (const float*)d_in[1];
  const float* points   = (const float*)d_in[2];
  const float* zero_tok = (const float*)d_in[3];
  const float* fn_g  = (const float*)d_in[4];
  const float* fn_b  = (const float*)d_in[5];
  const float* hn_g  = (const float*)d_in[6];
  const float* hn_b  = (const float*)d_in[7];
  const float* rp_w  = (const float*)d_in[8];
  const float* rp_b  = (const float*)d_in[9];
  const float* rpl_g = (const float*)d_in[10];
  const float* rpl_b = (const float*)d_in[11];
  const float* fe_w  = (const float*)d_in[12];
  const float* fe_b  = (const float*)d_in[13];
  const float* tn_g  = (const float*)d_in[14];
  const float* tn_b  = (const float*)d_in[15];
  const float* ln1_g = (const float*)d_in[16];
  const float* ln1_b = (const float*)d_in[17];
  const float* qkv_w = (const float*)d_in[18];
  const float* qkv_b = (const float*)d_in[19];
  const float* pr_w  = (const float*)d_in[20];
  const float* pr_b  = (const float*)d_in[21];
  const float* ln2_g = (const float*)d_in[22];
  const float* ln2_b = (const float*)d_in[23];
  const float* fc1_w = (const float*)d_in[24];
  const float* fc1_b = (const float*)d_in[25];
  const float* fc2_w = (const float*)d_in[26];
  const float* fc2_b = (const float*)d_in[27];
  const float* on_g  = (const float*)d_in[28];
  const float* on_b  = (const float*)d_in[29];
  float* out = (float*)d_out;

  // workspace carving (256B aligned)
  char* wp = (char*)d_ws;
  auto carve = [&](size_t bytes) -> char* {
    char* p = wp;
    wp += (bytes + 255) & ~(size_t)255;
    return p;
  };
  half_t* fe_w16  = (half_t*)carve((size_t)640 * 768 * 2);          // [768][640]
  half_t* qkv_w16 = (half_t*)carve((size_t)cL * 768 * 2304 * 2);    // per-l [2304][768]
  half_t* pr_w16  = (half_t*)carve((size_t)cL * 768 * 768 * 2);     // per-l [768][768]
  half_t* fc1_w16 = (half_t*)carve((size_t)cL * 768 * 3072 * 2);    // per-l [3072][768]
  half_t* fc2_w16 = (half_t*)carve((size_t)cL * 3072 * 768 * 2);    // per-l [768][3072]
  half_t* rp_w16  = (half_t*)carve((size_t)128 * 128 * 2);          // [128][128]
  half_t* hrm16   = (half_t*)carve((size_t)cM * 128 * 2);
  half_t* cat16   = (half_t*)carve((size_t)cM * 640 * 2);
  float*  rpo     = (float*) carve((size_t)cM * 128 * 4);
  float*  xbuf    = (float*) carve((size_t)cM * 768 * 4);
  float*  xs      = (float*) carve((size_t)cM * 768 * 4);
  half_t* h16     = (half_t*)carve((size_t)cM * 768 * 2);
  half_t* qkv16   = (half_t*)carve((size_t)cM * 2304 * 2);
  half_t* o16     = (half_t*)carve((size_t)cM * 768 * 2);
  half_t* hid16   = (half_t*)carve((size_t)cM * 3072 * 2);
  unsigned long long* keys = (unsigned long long*)carve((size_t)cB * cSORT * 8);
  int* gmin = (int*)carve(64);

  // weights: convert once to f16, transposed to [N][K]
  auto cvt_t = [&](const float* s, half_t* d, int K, int N) {
    cvt_f16_t_kernel<<<dim3(N / 32, K / 32), 256, 0, stream>>>(s, d, K, N);
  };
  cvt_t(fe_w, fe_w16, 640, 768);
  for (int l = 0; l < cL; ++l) {
    cvt_t(qkv_w + (size_t)l * 768 * 2304, qkv_w16 + (size_t)l * 2304 * 768, 768, 2304);
    cvt_t(pr_w  + (size_t)l * 768 * 768,  pr_w16  + (size_t)l * 768 * 768,  768, 768);
    cvt_t(fc1_w + (size_t)l * 768 * 3072, fc1_w16 + (size_t)l * 3072 * 768, 768, 3072);
    cvt_t(fc2_w + (size_t)l * 3072 * 768, fc2_w16 + (size_t)l * 768 * 3072, 3072, 768);
  }
  cvt_rp_pad_t_kernel<<<128, 128, 0, stream>>>(rp_w, rp_w16);

  // preprocessing
  prep_feats_kernel<<<cM, 256, 0, stream>>>(features, zero_tok, fn_g, fn_b, cat16);
  prep_hrm_kernel<<<cM, 128, 0, stream>>>(rgb, hn_g, hn_b, hrm16);
  gemm64_kernel<<<dim3(128 / 64, cM / 64), 128, 0, stream>>>(
      hrm16, rp_w16, rp_b, nullptr, rpo, nullptr, cM, 128, 128, 1 /*gelu*/);
  ln_f16_kernel<<<cM, 256, 0, stream>>>(rpo, 128, 128, rpl_g, rpl_b, cat16, 640, 0);
  gemm64_kernel<<<dim3(768 / 64, cM / 64), 128, 0, stream>>>(
      cat16, fe_w16, fe_b, nullptr, xbuf, nullptr, cM, 768, 640, 0);

  // morton keys + stable bitonic sort (64-bit key = code<<32 | index)
  init_min_kernel<<<1, 64, 0, stream>>>(gmin);
  vox_min_kernel<<<(cM + 255) / 256, 256, 0, stream>>>(points, gmin);
  make_keys_kernel<<<dim3(cSORT / 256, cB), 256, 0, stream>>>(points, gmin, keys);
  for (int k = 2; k <= cSORT; k <<= 1)
    for (int j = k >> 1; j > 0; j >>= 1)
      bitonic_pass_kernel<<<dim3(cSORT / 256, cB), 256, 0, stream>>>(keys, k, j);

  // gather into sorted order, then token norm (LN commutes with row permutation)
  gather_kernel<<<cM, 256, 0, stream>>>(xbuf, xs, keys);
  ln_f32_kernel<<<cM, 256, 0, stream>>>(xs, 768, tn_g, tn_b, xs);

  // transformer layers
  for (int l = 0; l < cL; ++l) {
    ln_f16_kernel<<<cM, 256, 0, stream>>>(xs, 768, 768, ln1_g + l * 768, ln1_b + l * 768,
                                          h16, 768, 0);
    gemm64_kernel<<<dim3(2304 / 64, cM / 64), 128, 0, stream>>>(
        h16, qkv_w16 + (size_t)l * 2304 * 768, qkv_b + l * 2304,
        nullptr, nullptr, qkv16, cM, 2304, 768, 0);
    attn_kernel<<<dim3(cH, cM / cPS), 256, ATTN_SMEM, stream>>>(qkv16, o16);
    gemm64_kernel<<<dim3(768 / 64, cM / 64), 128, 0, stream>>>(
        o16, pr_w16 + (size_t)l * 768 * 768, pr_b + l * 768,
        xs, xs, nullptr, cM, 768, 768, 0);
    ln_f16_kernel<<<cM, 256, 0, stream>>>(xs, 768, 768, ln2_g + l * 768, ln2_b + l * 768,
                                          h16, 768, 0);
    gemm64_kernel<<<dim3(3072 / 64, cM / 64), 128, 0, stream>>>(
        h16, fc1_w16 + (size_t)l * 3072 * 768, fc1_b + l * 3072,
        nullptr, nullptr, hid16, cM, 3072, 768, 1 /*gelu*/);
    gemm64_kernel<<<dim3(768 / 64, cM / 64), 128, 0, stream>>>(
        hid16, fc2_w16 + (size_t)l * 768 * 3072, fc2_b + l * 768,
        xs, xs, nullptr, cM, 768, 3072, 0);
  }

  // final LN + scatter through order (== take_along_axis with inverse perm)
  ln_scatter_kernel<<<cM, 256, 0, stream>>>(xs, on_g, on_b, keys, out);
}
